// TriangleAttention_72954314490348
// MI455X (gfx1250) — matrix operand order repaired and checked by reading, not compiled, not run on hardware
//
#include <hip/hip_runtime.h>
#include <math.h>

// ---------------------------------------------------------------------------
// TriangleAttention (starting node) for MI455X / gfx1250, wave32 + WMMA bf16.
//   N=384, C_IN=128, C_HID=32, H=4.  All GEMMs via v_wmma_f32_16x16x32_bf16.
// ---------------------------------------------------------------------------

typedef __attribute__((ext_vector_type(16))) __bf16 v16bf;
typedef __attribute__((ext_vector_type(8)))  float  v8f;
typedef int v4i __attribute__((vector_size(16)));
typedef __attribute__((address_space(1))) v4i* gv4i_p;   // global v4i*
typedef __attribute__((address_space(3))) v4i* lv4i_p;   // LDS v4i*

#define NSEQ 384
#define NPOS (384 * 384)
#define CIN  128
#define QSCALE 0.17677669529663689f   // 1/sqrt(32)

#if defined(__has_builtin)
#  if __has_builtin(__builtin_amdgcn_global_load_async_to_lds_b128)
#    define HAVE_ASYNC_LDS 1
#  endif
#  if __has_builtin(__builtin_amdgcn_s_wait_asynccnt)
#    define HAVE_WAIT_ASYNC 1
#  endif
#endif

__device__ __forceinline__ v8f wmma_bf16(v16bf a, v16bf b, v8f c) {
  return __builtin_amdgcn_wmma_f32_16x16x32_bf16(false, a, false, b, (short)0, c,
                                                 false, false);
}

// A fragment (16x32 bf16) from row-major [row][k] with row stride rs.
// Contiguous-k layout used here (B matches): lane&15 = row M; element e: K = (lane>>4)*16 + e
// -> per lane: two contiguous 16B chunks at k = hi*16 and k = hi*16 + 8.
__device__ __forceinline__ v16bf load_a_frag(const __bf16* src, int rs) {
  int lane = threadIdx.x & 31;
  int m = lane & 15, hi = lane >> 4;
  const __bf16* p = src + m * rs + hi * 16;
  union { uint4 u[2]; v16bf v; } u;
  u.u[0] = *(const uint4*)(p);
  u.u[1] = *(const uint4*)(p + 8);
  return u.v;
}

// B fragment (32x16), B[k][n] = src[n*ns + k] (k contiguous in memory).
// Contiguous-k layout used here (A matches): lane&15 = col N; element e: K = (lane>>4)*16 + e
// -> per lane: two contiguous 16B chunks at k = hi*16 and k = hi*16 + 8.
__device__ __forceinline__ v16bf load_b_kcont(const __bf16* src, int ns) {
  int lane = threadIdx.x & 31;
  int n = lane & 15, hi = lane >> 4;
  const __bf16* p = src + n * ns + (hi << 4);
  union { uint4 u[2]; v16bf v; } u;
  u.u[0] = *(const uint4*)(p);
  u.u[1] = *(const uint4*)(p + 8);
  return u.v;
}

// B fragment from f32 weights W[out][in] (row stride 128): B[k][n]=W[nb+n][kb+k]
__device__ __forceinline__ v16bf load_b_wf(const float* W, int n_base, int k_base) {
  int lane = threadIdx.x & 31;
  int n = lane & 15, hi = lane >> 4;
  const float4* p =
      (const float4*)(W + (size_t)(n_base + n) * CIN + k_base + (hi << 4));
  v16bf b;
#pragma unroll
  for (int q4 = 0; q4 < 4; ++q4) {
    float4 f = p[q4];
    b[q4 * 4 + 0] = (__bf16)f.x;
    b[q4 * 4 + 1] = (__bf16)f.y;
    b[q4 * 4 + 2] = (__bf16)f.z;
    b[q4 * 4 + 3] = (__bf16)f.w;
  }
  return b;
}

// ---------------------------------------------------------------------------
// K1: LayerNorm + q/k/v/g/tri-bias projections.
//   block = 256 thr (8 waves), 64 positions (one row segment) per block.
//   n-tiles: 0-7 q, 8-15 k, 16-23 v, 24-31 g, 32 = tri-bias (4 valid ch).
// ---------------------------------------------------------------------------
__global__ __launch_bounds__(256) void k1_ln_proj(
    const float* __restrict__ x, const float* __restrict__ gamma,
    const float* __restrict__ beta, const float* __restrict__ wbias,
    const float* __restrict__ wq, const float* __restrict__ wk,
    const float* __restrict__ wv, const float* __restrict__ wg,
    const float* __restrict__ bg, void* qb_, void* kb_, void* vb_, void* gb_,
    float* __restrict__ tb) {
  __bf16* qb = (__bf16*)qb_;
  __bf16* kb = (__bf16*)kb_;
  __bf16* vb = (__bf16*)vb_;
  __bf16* gb = (__bf16*)gb_;

  __shared__ __align__(16) __bf16 xn_s[64 * CIN];  // 16 KB

  const int P0 = blockIdx.x * 64;
  const int t = threadIdx.x;
  const int p = t >> 2, part = t & 3;

  // --- LayerNorm: 4 threads per position, 32 channels each (float4 I/O) ---
  const float4* xp4 = (const float4*)(x + (size_t)(P0 + p) * CIN + part * 32);
  float4 xr[8];
  float s1 = 0.f, s2 = 0.f;
#pragma unroll
  for (int u = 0; u < 8; ++u) {
    float4 f = xp4[u];
    xr[u] = f;
    s1 += f.x + f.y + f.z + f.w;
    s2 += f.x * f.x + f.y * f.y + f.z * f.z + f.w * f.w;
  }
  s1 += __shfl_xor(s1, 1); s1 += __shfl_xor(s1, 2);
  s2 += __shfl_xor(s2, 1); s2 += __shfl_xor(s2, 2);
  float mu = s1 * (1.f / 128.f);
  float var = s2 * (1.f / 128.f) - mu * mu;
  float rstd = rsqrtf(var + 1e-5f);
#pragma unroll
  for (int u = 0; u < 4; ++u) {  // two float4 -> one uint4 (8 bf16) per step
    union { uint4 u4; __bf16 h[8]; } pk;
#pragma unroll
    for (int e = 0; e < 8; ++e) {
      int c = u * 8 + e;
      int ch = part * 32 + c;
      float xv = (&xr[c >> 2].x)[c & 3];
      pk.h[e] = (__bf16)((xv - mu) * rstd * gamma[ch] + beta[ch]);
    }
    ((uint4*)(xn_s + p * CIN + part * 32))[u] = pk.u4;
  }
  __syncthreads();

  // --- WMMA GEMM: [64 x 128] x [128 x 516] ---
  const int wid = t >> 5, lane = t & 31;
  const int ln16 = lane & 15, hi = lane >> 4;
  const int i_row = P0 / NSEQ;             // block-uniform (64 | 384)
  const int j_base = P0 - i_row * NSEQ;

  for (int tile = wid; tile < 4 * 33; tile += 8) {
    int mt = tile & 3;
    int nt = tile >> 2;
    v8f acc = {0.f, 0.f, 0.f, 0.f, 0.f, 0.f, 0.f, 0.f};
#pragma unroll
    for (int kk = 0; kk < 4; ++kk) {
      v16bf a = load_a_frag(xn_s + mt * 16 * CIN + kk * 32, CIN);
      v16bf b;
      if (nt < 32) {
        const float* W = (nt < 8) ? wq : (nt < 16) ? wk : (nt < 24) ? wv : wg;
        b = load_b_wf(W, (nt & 7) * 16, kk * 32);
      } else {
        // tri-bias projection: clamp row (branch-free), select-to-zero
        int nc = (ln16 < 4) ? ln16 : 0;
        const float4* p =
            (const float4*)(wbias + nc * CIN + kk * 32 + (hi << 4));
#pragma unroll
        for (int q4 = 0; q4 < 4; ++q4) {
          float4 f = p[q4];
          bool ok = (ln16 < 4);
          b[q4 * 4 + 0] = ok ? (__bf16)f.x : (__bf16)0.f;
          b[q4 * 4 + 1] = ok ? (__bf16)f.y : (__bf16)0.f;
          b[q4 * 4 + 2] = ok ? (__bf16)f.z : (__bf16)0.f;
          b[q4 * 4 + 3] = ok ? (__bf16)f.w : (__bf16)0.f;
        }
      }
      acc = wmma_bf16(a, b, acc);
    }
    // epilogue: C/D layout -> M = r + 8*hi, N = ln16
#pragma unroll
    for (int r = 0; r < 8; ++r) {
      int j = j_base + mt * 16 + r + hi * 8;
      int P = P0 + mt * 16 + r + hi * 8;
      if (nt < 24) {
        int o = (nt & 7) * 16 + ln16;
        int h = o >> 5, d = o & 31;
        size_t idx = ((size_t)(i_row * 4 + h) * NSEQ + j) * 32 + d;
        float val = acc[r];
        if (nt < 8)       qb[idx] = (__bf16)(val * QSCALE);
        else if (nt < 16) kb[idx] = (__bf16)val;
        else              vb[idx] = (__bf16)val;
      } else if (nt < 32) {
        int o = (nt - 24) * 16 + ln16;
        float val = acc[r] + bg[o];
        gb[(size_t)P * CIN + o] = (__bf16)(1.0f / (1.0f + expf(-val)));
      } else if (ln16 < 4) {
        tb[(size_t)ln16 * NPOS + P] = acc[r];
      }
    }
  }
}

// ---------------------------------------------------------------------------
// K2: flash attention per (row i, head h).  384 q x 384 k, Dh=32.
//   K staged via async global->LDS (ASYNCcnt); V staged transposed so both
//   B fragments are contiguous ds_load_b128; Q A-fragments straight from
//   global.  Online softmax; P relayout via per-wave LDS scratch + dscnt.
// ---------------------------------------------------------------------------
__global__ __launch_bounds__(256) void k2_attn(
    const void* qb_, const void* kb_, const void* vb_, const void* gb_,
    const float* __restrict__ tb, const float* __restrict__ mask, void* go_) {
  const __bf16* qb = (const __bf16*)qb_;
  const __bf16* kb = (const __bf16*)kb_;
  const __bf16* vb = (const __bf16*)vb_;
  const __bf16* gb = (const __bf16*)gb_;
  __bf16* go = (__bf16*)go_;

  __shared__ __align__(16) __bf16 k_s[NSEQ * 32];    // 24 KB [j'][d]
  __shared__ __align__(16) __bf16 vt_s[32 * NSEQ];   // 24 KB [d][j'] (transposed)
  __shared__ float mb_s[NSEQ];
  __shared__ __align__(16) __bf16 pscr[8][16 * 32];  // 8 KB P scratch

  const int I = blockIdx.x >> 2;
  const int h = blockIdx.x & 3;
  const int t = threadIdx.x;
  const size_t base = (size_t)(I * 4 + h) * (NSEQ * 32);

  // ---- stage K tile (async DMA to LDS when available) ----
  {
    const __bf16* kg = kb + base;
#if HAVE_ASYNC_LDS
#pragma unroll
    for (int u = 0; u < 6; ++u) {
      int idx = t + u * 256;
      __builtin_amdgcn_global_load_async_to_lds_b128(
          (gv4i_p)(unsigned long long)(kg + idx * 8),
          (lv4i_p)(unsigned int)(unsigned long long)(k_s + idx * 8), 0, 0);
    }
#else
    const uint4* ks4 = (const uint4*)kg;
    uint4* kd4 = (uint4*)k_s;
#pragma unroll
    for (int u = 0; u < 6; ++u) {
      int idx = t + u * 256;
      kd4[idx] = ks4[idx];
    }
#endif
  }
  // ---- stage V transposed ----
  {
    const uint4* vg4 = (const uint4*)(vb + base);
#pragma unroll
    for (int u = 0; u < 6; ++u) {
      int idx = t + u * 256;
      union { uint4 u4; __bf16 hh[8]; } pk;
      pk.u4 = vg4[idx];
      int j = idx >> 2;
      int d0 = (idx & 3) * 8;
#pragma unroll
      for (int e = 0; e < 8; ++e) vt_s[(d0 + e) * NSEQ + j] = pk.hh[e];
    }
  }
  for (int u = t; u < NSEQ; u += 256)
    mb_s[u] = (mask[I * NSEQ + u] - 1.0f) * 1e9f;

#if HAVE_ASYNC_LDS
#  if HAVE_WAIT_ASYNC
  __builtin_amdgcn_s_wait_asynccnt(0);
#  else
  asm volatile("s_wait_asynccnt 0" ::: "memory");
#  endif
#endif
  __syncthreads();

  const int wid = t >> 5, lane = t & 31;
  const int ln16 = lane & 15, hi = lane >> 4;
  const float* tbh = tb + (size_t)h * NPOS;

  for (int mi = 0; mi < 3; ++mi) {
    int m = wid + mi * 8;
    v16bf qa = load_a_frag(qb + base + m * 16 * 32, 32);  // Q direct from global

    float rmax[8], rsum[8];
#pragma unroll
    for (int r = 0; r < 8; ++r) { rmax[r] = -3.0e38f; rsum[r] = 0.f; }
    v8f o0 = {0.f, 0.f, 0.f, 0.f, 0.f, 0.f, 0.f, 0.f};
    v8f o1 = {0.f, 0.f, 0.f, 0.f, 0.f, 0.f, 0.f, 0.f};

    for (int n2 = 0; n2 < 24; n2 += 2) {
      v8f z = {0.f, 0.f, 0.f, 0.f, 0.f, 0.f, 0.f, 0.f};
      // S = Q*K^T : B[k=d][n=j'] = k_s[j'*32 + d] (k contiguous)
      v16bf bk0 = load_b_kcont(k_s + (n2 + 0) * 16 * 32, 32);
      v8f s0 = wmma_bf16(qa, bk0, z);
      v16bf bk1 = load_b_kcont(k_s + (n2 + 1) * 16 * 32, 32);
      v8f s1 = wmma_bf16(qa, bk1, z);

      // + tri_bias[h, q, k] + mask_bias[k]
#pragma unroll
      for (int r = 0; r < 8; ++r) {
        int qi = m * 16 + r + hi * 8;
        int j0 = n2 * 16 + ln16;
        s0[r] += tbh[(size_t)qi * NSEQ + j0] + mb_s[j0];
        s1[r] += tbh[(size_t)qi * NSEQ + j0 + 16] + mb_s[j0 + 16];
      }

      // online softmax over the 32 new columns
#pragma unroll
      for (int r = 0; r < 8; ++r) {
        float tmax = fmaxf(s0[r], s1[r]);
        tmax = fmaxf(tmax, __shfl_xor(tmax, 1));
        tmax = fmaxf(tmax, __shfl_xor(tmax, 2));
        tmax = fmaxf(tmax, __shfl_xor(tmax, 4));
        tmax = fmaxf(tmax, __shfl_xor(tmax, 8));
        float nm = fmaxf(rmax[r], tmax);
        float corr = expf(rmax[r] - nm);
        float p0 = expf(s0[r] - nm);
        float p1 = expf(s1[r] - nm);
        float ts = p0 + p1;
        ts += __shfl_xor(ts, 1);
        ts += __shfl_xor(ts, 2);
        ts += __shfl_xor(ts, 4);
        ts += __shfl_xor(ts, 8);
        rsum[r] = rsum[r] * corr + ts;
        rmax[r] = nm;
        o0[r] *= corr;
        o1[r] *= corr;
        int row = r + hi * 8;  // C-layout -> row-major scratch
        pscr[wid][row * 32 + ln16] = (__bf16)p0;
        pscr[wid][row * 32 + 16 + ln16] = (__bf16)p1;
      }
      // per-wave LDS RAW: wait for DS stores before re-reading as A fragment
      asm volatile("s_wait_dscnt 0" ::: "memory");
      v16bf pa = load_a_frag(&pscr[wid][0], 32);

      // O += P*V : B[k=j'][n=d] = vt_s[d*384 + j'] (k contiguous)
      v16bf bv0 = load_b_kcont(vt_s + n2 * 16, NSEQ);
      v16bf bv1 = load_b_kcont(vt_s + 16 * NSEQ + n2 * 16, NSEQ);
      o0 = wmma_bf16(pa, bv0, o0);
      o1 = wmma_bf16(pa, bv1, o1);
    }

    // normalize, gate, store gated output (bf16) in [pos][h*32+d] layout
#pragma unroll
    for (int r = 0; r < 8; ++r) {
      int qi = m * 16 + r + hi * 8;
      float inv = rsum[r] > 0.f ? 1.0f / rsum[r] : 0.f;
      size_t gbase = ((size_t)I * NSEQ + qi) * CIN + h * 32;
      float g0 = (float)gb[gbase + ln16];
      float g1 = (float)gb[gbase + 16 + ln16];
      go[gbase + ln16]      = (__bf16)(o0[r] * inv * g0);
      go[gbase + 16 + ln16] = (__bf16)(o1[r] * inv * g1);
    }
  }
}

// ---------------------------------------------------------------------------
// K3: output projection  out[P][c] = sum_k go[P][k] * wo[c][k] + bo[c]
// ---------------------------------------------------------------------------
__global__ __launch_bounds__(256) void k3_out(const void* go_,
                                              const float* __restrict__ wo,
                                              const float* __restrict__ bo,
                                              float* __restrict__ out) {
  const __bf16* go = (const __bf16*)go_;
  __shared__ __align__(16) __bf16 a_s[64 * CIN];  // 16 KB

  const int P0 = blockIdx.x * 64;
  const int t = threadIdx.x;
  {
    const uint4* src = (const uint4*)(go + (size_t)P0 * CIN);
    uint4* dst = (uint4*)a_s;
#pragma unroll
    for (int u = 0; u < 4; ++u) dst[t + u * 256] = src[t + u * 256];
  }
  __syncthreads();

  const int wid = t >> 5, lane = t & 31;
  const int ln16 = lane & 15, hi = lane >> 4;

  for (int tile = wid; tile < 32; tile += 8) {
    int mt = tile & 3, nt = tile >> 2;
    v8f acc = {0.f, 0.f, 0.f, 0.f, 0.f, 0.f, 0.f, 0.f};
#pragma unroll
    for (int kk = 0; kk < 4; ++kk) {
      v16bf a = load_a_frag(a_s + mt * 16 * CIN + kk * 32, CIN);
      v16bf b = load_b_wf(wo, nt * 16, kk * 32);
      acc = wmma_bf16(a, b, acc);
    }
#pragma unroll
    for (int r = 0; r < 8; ++r) {
      int P = P0 + mt * 16 + r + hi * 8;
      int c = nt * 16 + ln16;
      out[(size_t)P * CIN + c] = acc[r] + bo[c];
    }
  }
}

// ---------------------------------------------------------------------------
extern "C" void kernel_launch(void* const* d_in, const int* in_sizes, int n_in,
                              void* d_out, int out_size, void* d_ws,
                              size_t ws_size, hipStream_t stream) {
  const float* x     = (const float*)d_in[0];
  const float* mask  = (const float*)d_in[1];
  const float* gamma = (const float*)d_in[2];
  const float* beta  = (const float*)d_in[3];
  const float* wbias = (const float*)d_in[4];
  const float* wq    = (const float*)d_in[5];
  const float* wk    = (const float*)d_in[6];
  const float* wv    = (const float*)d_in[7];
  const float* wg    = (const float*)d_in[8];
  const float* bg    = (const float*)d_in[9];
  const float* wo    = (const float*)d_in[10];
  const float* bo    = (const float*)d_in[11];
  float* out = (float*)d_out;

  const size_t E = (size_t)NPOS * CIN;  // elements per bf16 buffer (x2 bytes)
  char* w = (char*)d_ws;
  void* qb = (void*)(w);
  void* kb = (void*)(w + 2 * E);
  void* vb = (void*)(w + 4 * E);
  void* gb = (void*)(w + 6 * E);
  void* go = (void*)(w + 8 * E);
  float* tb = (float*)(w + 10 * E);  // 4 * NPOS floats

  k1_ln_proj<<<NPOS / 64, 256, 0, stream>>>(x, gamma, beta, wbias, wq, wk, wv,
                                            wg, bg, qb, kb, vb, gb, tb);
  k2_attn<<<NSEQ * 4, 256, 0, stream>>>(qb, kb, vb, gb, tb, mask, go);
  k3_out<<<NPOS / 64, 256, 0, stream>>>(go, wo, bo, out);
}
